// BuildMoE_88931592831123
// MI455X (gfx1250) — compile-verified, matrix operand order checked by reading, not run on hardware
//
#include <hip/hip_runtime.h>

typedef __attribute__((ext_vector_type(16))) __bf16 v16bf;
typedef __attribute__((ext_vector_type(8)))  float v8f;

// Problem constants (fixed reference config: B,T,D,E,L,MAXP,PRED,TOPK)
constexpr int B = 128, T = 2048, D = 256, E = 32, L = 8, MAXP = 168, P = 168;
constexpr float SCALE = 0.0625f; // 1/sqrt(256)

// Workspace layout (float offsets)
constexpr size_t OFF_STEPBASE = 0;                              // 168*256
constexpr size_t OFF_Q1       = OFF_STEPBASE + (size_t)P * D;   // 8*256
constexpr size_t OFF_QK       = OFF_Q1 + (size_t)L * D;         // 8*256 (scaled)
constexpr size_t OFF_Q2S      = OFF_QK + (size_t)L * D;         // 168*256 (scaled)
constexpr size_t OFF_WCVLO    = OFF_Q2S + (size_t)P * D;        // 256*256 f32
constexpr size_t OFF_WSOGB    = OFF_WCVLO + (size_t)D * D;      // 256*256 bf16, [N][K] transposed
constexpr size_t OFF_GBASE    = OFF_WSOGB + (size_t)D * D / 2;  // 168*256
constexpr int    FLASH_REC    = 16 + L * D;                     // m[8], l[8], acc[8][256]
constexpr size_t OFF_FLASH    = OFF_GBASE + (size_t)P * D;      // B*2*FLASH_REC
constexpr size_t OFF_K2       = OFF_FLASH + (size_t)B * 2 * FLASH_REC;
constexpr size_t OFF_V2       = OFF_K2 + (size_t)B * L * D;

// ---------------------------------------------------------------------------
// K0a: step_base, q1 = latents@w_lat_q, W_cv_lo = w_ctx_v@w_lat_out,
//      W_so_g^T (bf16) = (w_step_out@gate_w1[256:])^T
__global__ void __launch_bounds__(256) k0a(const float* __restrict__ qpos,
                                           const float* __restrict__ lemb,
                                           const float* __restrict__ lat,
                                           const float* __restrict__ wlq,
                                           const float* __restrict__ wcv,
                                           const float* __restrict__ wlo,
                                           const float* __restrict__ wso,
                                           const float* __restrict__ gw1,
                                           float* __restrict__ ws) {
  int bid = blockIdx.x, tid = threadIdx.x;
  if (bid < P) {
    ws[OFF_STEPBASE + (size_t)bid * D + tid] = qpos[(size_t)bid * D + tid] + lemb[(size_t)P * D + tid];
  } else if (bid < P + L) {
    int i = bid - P; float acc = 0.f;
    for (int e = 0; e < D; ++e) acc += lat[i * D + e] * wlq[e * D + tid];
    ws[OFF_Q1 + (size_t)i * D + tid] = acc;
  } else if (bid < P + L + D) {
    int e = bid - (P + L); float acc = 0.f;
    for (int m = 0; m < D; ++m) acc += wcv[e * D + m] * wlo[m * D + tid];
    ws[OFF_WCVLO + (size_t)e * D + tid] = acc;
  } else {
    int e = bid - (P + L + D); float acc = 0.f;  // e = K index
    for (int m = 0; m < D; ++m) acc += wso[e * D + m] * gw1[(size_t)(D + m) * D + tid];
    // store transposed [n][k] in bf16
    ((__bf16*)(ws + OFF_WSOGB))[(size_t)tid * D + e] = (__bf16)acc;
  }
}

// K0b: qk = (q1 @ w_ctx_k^T)*scale, q2s = (step_base@w_step_q)*scale,
//      g_base = step_base@gate_w1[:256] + gate_b1
__global__ void __launch_bounds__(256) k0b(const float* __restrict__ wck,
                                           const float* __restrict__ wsq,
                                           const float* __restrict__ gw1,
                                           const float* __restrict__ gb1,
                                           float* __restrict__ ws) {
  const float* sb = ws + OFF_STEPBASE;
  const float* q1 = ws + OFF_Q1;
  int bid = blockIdx.x, tid = threadIdx.x;
  if (bid < L) {
    float acc = 0.f;
    for (int d = 0; d < D; ++d) acc += q1[bid * D + d] * wck[(size_t)tid * D + d];
    ws[OFF_QK + (size_t)bid * D + tid] = acc * SCALE;
  } else if (bid < L + P) {
    int p = bid - L; float acc = 0.f;
    for (int e = 0; e < D; ++e) acc += sb[p * D + e] * wsq[e * D + tid];
    ws[OFF_Q2S + (size_t)p * D + tid] = acc * SCALE;
  } else {
    int p = bid - (L + P); float acc = gb1[tid];
    for (int e = 0; e < D; ++e) acc += sb[p * D + e] * gw1[(size_t)e * D + tid];
    ws[OFF_GBASE + (size_t)p * D + tid] = acc;
  }
}

// ---------------------------------------------------------------------------
// K1: flash attention over ctx. grid = B*2 (batch, T-chunk). 8 waves/WG; each
// wave holds qk (8x8 slice/lane) + acc[8][8] in registers; streams 128 rows.
__global__ void __launch_bounds__(256) k1(const float* __restrict__ ctx,
                                          float* __restrict__ ws) {
  const float* qk = ws + OFF_QK;
  float* flash = ws + OFF_FLASH;
  const int b = blockIdx.x >> 1, c = blockIdx.x & 1;
  const int tid = threadIdx.x, w = tid >> 5, lane = tid & 31;

  float qkr[L][8];
#pragma unroll
  for (int i = 0; i < L; ++i) {
    float4 x0 = *(const float4*)(qk + i * D + lane * 8);
    float4 x1 = *(const float4*)(qk + i * D + lane * 8 + 4);
    qkr[i][0] = x0.x; qkr[i][1] = x0.y; qkr[i][2] = x0.z; qkr[i][3] = x0.w;
    qkr[i][4] = x1.x; qkr[i][5] = x1.y; qkr[i][6] = x1.z; qkr[i][7] = x1.w;
  }
  float acc[L][8], m[L], lsum[L];
#pragma unroll
  for (int i = 0; i < L; ++i) {
    m[i] = -INFINITY; lsum[i] = 0.f;
#pragma unroll
    for (int dd = 0; dd < 8; ++dd) acc[i][dd] = 0.f;
  }
  const float* cb = ctx + (size_t)b * T * D;
  const int t0 = c * (T / 2) + w * (T / 16); // 128 rows per wave

  for (int g = 0; g < T / 16; g += 4) {
    float r[4][8];
#pragma unroll
    for (int rr = 0; rr < 4; ++rr) {
      const float* rp = cb + (size_t)(t0 + g + rr) * D + lane * 8;
      float4 a0 = *(const float4*)rp;
      float4 a1 = *(const float4*)(rp + 4);
      r[rr][0] = a0.x; r[rr][1] = a0.y; r[rr][2] = a0.z; r[rr][3] = a0.w;
      r[rr][4] = a1.x; r[rr][5] = a1.y; r[rr][6] = a1.z; r[rr][7] = a1.w;
    }
#pragma unroll
    for (int i = 0; i < L; ++i) {
      float s0 = 0, s1 = 0, s2 = 0, s3 = 0;
#pragma unroll
      for (int dd = 0; dd < 8; ++dd) {
        s0 += qkr[i][dd] * r[0][dd]; s1 += qkr[i][dd] * r[1][dd];
        s2 += qkr[i][dd] * r[2][dd]; s3 += qkr[i][dd] * r[3][dd];
      }
#pragma unroll
      for (int off = 16; off >= 1; off >>= 1) {
        s0 += __shfl_xor(s0, off); s1 += __shfl_xor(s1, off);
        s2 += __shfl_xor(s2, off); s3 += __shfl_xor(s3, off);
      }
      float mt = fmaxf(fmaxf(s0, s1), fmaxf(s2, s3));
      float mn = fmaxf(m[i], mt);
      float sc = __expf(m[i] - mn);
      float p0 = __expf(s0 - mn), p1 = __expf(s1 - mn);
      float p2 = __expf(s2 - mn), p3 = __expf(s3 - mn);
      lsum[i] = lsum[i] * sc + (p0 + p1 + p2 + p3);
#pragma unroll
      for (int dd = 0; dd < 8; ++dd)
        acc[i][dd] = acc[i][dd] * sc + p0 * r[0][dd] + p1 * r[1][dd] + p2 * r[2][dd] + p3 * r[3][dd];
      m[i] = mn;
    }
  }
  __shared__ float red[8][FLASH_REC];
#pragma unroll
  for (int i = 0; i < L; ++i) {
#pragma unroll
    for (int dd = 0; dd < 8; ++dd) red[w][16 + i * D + lane * 8 + dd] = acc[i][dd];
    if (lane == 0) { red[w][i] = m[i]; red[w][8 + i] = lsum[i]; }
  }
  __syncthreads();
  // combine 8 waves -> one chunk record
  const int i = tid >> 5, d0 = (tid & 31) * 8;
  float mst = -INFINITY;
  for (int ww = 0; ww < 8; ++ww) mst = fmaxf(mst, red[ww][i]);
  float al[8], lst = 0.f;
  for (int ww = 0; ww < 8; ++ww) { al[ww] = __expf(red[ww][i] - mst); lst += red[ww][8 + i] * al[ww]; }
  float* rec = flash + (size_t)(b * 2 + c) * FLASH_REC;
  for (int dd = 0; dd < 8; ++dd) {
    float s = 0.f;
    for (int ww = 0; ww < 8; ++ww) s += red[ww][16 + i * D + d0 + dd] * al[ww];
    rec[16 + i * D + d0 + dd] = s;
  }
  if ((tid & 31) == 0) { rec[i] = mst; rec[8 + i] = lst; }
}

// ---------------------------------------------------------------------------
// K2: combine 2 chunks -> lat_pre; lat_ctx = lat_pre @ W_cv_lo; k2/v2 proj.
__global__ void __launch_bounds__(256) k2(const float* __restrict__ wlk,
                                          const float* __restrict__ wlv,
                                          float* __restrict__ ws) {
  const int b = blockIdx.x, tid = threadIdx.x;
  const float* r0 = ws + OFF_FLASH + (size_t)(b * 2) * FLASH_REC;
  const float* r1 = r0 + FLASH_REC;
  const float* wcvlo = ws + OFF_WCVLO;
  __shared__ float lp[L * D];
  __shared__ float lc[L * D];
  {
    int i = tid >> 5, d0 = (tid & 31) * 8;
    float m0 = r0[i], m1 = r1[i], mm = fmaxf(m0, m1);
    float a0 = __expf(m0 - mm), a1 = __expf(m1 - mm);
    float ll = r0[8 + i] * a0 + r1[8 + i] * a1;
    for (int dd = 0; dd < 8; ++dd) {
      int d = d0 + dd;
      lp[i * D + d] = (r0[16 + i * D + d] * a0 + r1[16 + i * D + d] * a1) / ll;
    }
  }
  __syncthreads();
  {
    float a[L] = {0};
    for (int e = 0; e < D; ++e) {
      float wv = wcvlo[(size_t)e * D + tid];
#pragma unroll
      for (int i = 0; i < L; ++i) a[i] += lp[i * D + e] * wv;
    }
#pragma unroll
    for (int i = 0; i < L; ++i) lc[i * D + tid] = a[i];
  }
  __syncthreads();
  {
    float ak[L] = {0}, av[L] = {0};
    for (int e = 0; e < D; ++e) {
      float wk = wlk[(size_t)e * D + tid], wv = wlv[(size_t)e * D + tid];
#pragma unroll
      for (int i = 0; i < L; ++i) { float x = lc[i * D + e]; ak[i] += x * wk; av[i] += x * wv; }
    }
#pragma unroll
    for (int i = 0; i < L; ++i) {
      ws[OFF_K2 + (size_t)b * L * D + i * D + tid] = ak[i];
      ws[OFF_V2 + (size_t)b * L * D + i * D + tid] = av[i];
    }
  }
}

// ---------------------------------------------------------------------------
// K3: per (batch, 64-row p-tile): step attention (8 keys), WMMA bf16 GEMM
// attn_out @ W_so_g, GELU gate, logits, top-2 softmax, expert mix.
__global__ void __launch_bounds__(256) k3(const float* __restrict__ ws,
                                          const float* __restrict__ gw2,
                                          const float* __restrict__ gb2,
                                          const float* __restrict__ qexp,
                                          float* __restrict__ out) {
  const int b = blockIdx.y, p0 = blockIdx.x * 64, tid = threadIdx.x;
  const float* q2s = ws + OFF_Q2S;
  const float* gbse = ws + OFF_GBASE;
  const float* k2g = ws + OFF_K2 + (size_t)b * L * D;
  const float* v2g = ws + OFF_V2 + (size_t)b * L * D;

  __shared__ float sA[64 * 257];     // q2s tile, later H (gate pre-act / h)
  __shared__ float k2s[L * 264];
  __shared__ float v2s[L * 264];
  __shared__ float a_s[64 * 8];
  __shared__ __align__(16) __bf16 attnB[64 * 264];
  __shared__ float lg[64 * E];

  for (int idx = tid; idx < 64 * D; idx += 256) {
    int r = idx >> 8, d = idx & 255;
    int rc = (p0 + r < P) ? (p0 + r) : (P - 1);
    float v = q2s[(size_t)rc * D + d];
    sA[r * 257 + d] = (p0 + r < P) ? v : 0.f;
  }
  for (int idx = tid; idx < L * D; idx += 256) {
    int r = idx >> 8, d = idx & 255;
    k2s[r * 264 + d] = k2g[idx];
    v2s[r * 264 + d] = v2g[idx];
  }
  __syncthreads();

  { // Phase A: scores (p x 8) + softmax over j
    int w = tid >> 5, lane = tid & 31, j = lane & 7;
#pragma unroll
    for (int q = 0; q < 2; ++q) {
      int pl = w * 8 + q * 4 + (lane >> 3);
      float s = 0.f;
#pragma unroll 4
      for (int d = 0; d < D; ++d) s += sA[pl * 257 + d] * k2s[j * 264 + d];
      float mx = s;
      mx = fmaxf(mx, __shfl_xor(mx, 1));
      mx = fmaxf(mx, __shfl_xor(mx, 2));
      mx = fmaxf(mx, __shfl_xor(mx, 4));
      float e = __expf(s - mx);
      float sm = e;
      sm += __shfl_xor(sm, 1); sm += __shfl_xor(sm, 2); sm += __shfl_xor(sm, 4);
      a_s[pl * 8 + j] = e / sm;
    }
  }
  __syncthreads();

  { // attn_out -> bf16 tile
    int pl = tid >> 2, q = tid & 3;
    float aj[8];
#pragma unroll
    for (int j = 0; j < 8; ++j) aj[j] = a_s[pl * 8 + j];
    for (int dd = 0; dd < 64; ++dd) {
      int d = q * 64 + dd;
      float s = 0.f;
#pragma unroll
      for (int j = 0; j < 8; ++j) s += aj[j] * v2s[j * 264 + d];
      attnB[pl * 264 + d] = (__bf16)s;
    }
  }
  __syncthreads();

  { // Phase B: H(64x256) = attnB @ W_so_g via v_wmma_f32_16x16x32_bf16
    // A fragments preloaded (b128 LDS pairs), B from pre-transposed bf16
    // global (b128 pairs). Two accumulators break the WMMA RAW chain.
    int w = tid >> 5, lane = tid & 31;
    int m_t = w & 3;
    int nbase = (w >> 2) * 8;
    int row = m_t * 16 + (lane & 15);
    int kh = (lane >> 4);            // 0 or 1 (K-half select), khalf = kh*8
    const __bf16* wsogB = (const __bf16*)(ws + OFF_WSOGB);
    union U { v16bf v; uint4 q[2]; };
    U Afr[8];
#pragma unroll
    for (int kt = 0; kt < 8; ++kt) {
      const uint4* ap = (const uint4*)(attnB + row * 264 + kt * 32 + kh * 8);
      Afr[kt].q[0] = ap[0];   // K = k0 .. k0+7
      Afr[kt].q[1] = ap[2];   // K = k0+16 .. k0+23
    }
#pragma unroll 1
    for (int nn = 0; nn < 8; ++nn) {
      int ncol = (nbase + nn) * 16 + (lane & 15);
      const uint4* bp = (const uint4*)(wsogB + (size_t)ncol * D);
      v8f c0 = {}, c1 = {};
#pragma unroll
      for (int kt = 0; kt < 8; kt += 2) {
        U b0, b1;
        const uint4* bp0 = bp + kt * 4 + kh;
        const uint4* bp1 = bp + (kt + 1) * 4 + kh;
        b0.q[0] = bp0[0]; b0.q[1] = bp0[2];
        b1.q[0] = bp1[0]; b1.q[1] = bp1[2];
        c0 = __builtin_amdgcn_wmma_f32_16x16x32_bf16(false, Afr[kt].v, false, b0.v,
                                                     (short)0, c0, false, false);
        c1 = __builtin_amdgcn_wmma_f32_16x16x32_bf16(false, Afr[kt + 1].v, false, b1.v,
                                                     (short)0, c1, false, false);
      }
      int rbase = m_t * 16 + kh * 8;
#pragma unroll
      for (int v = 0; v < 8; ++v) sA[(rbase + v) * 257 + ncol] = c0[v] + c1[v];
    }
  }
  __syncthreads();

  // Phase C: h = gelu(g_base + H) (exact erf gelu), in place
  for (int k = 0; k < 64; ++k) {
    int idx = tid + k * 256;
    int r = idx >> 8, d = idx & 255;
    int rc = (p0 + r < P) ? (p0 + r) : (P - 1);   // clamped, unconditional load
    float x = sA[r * 257 + d] + gbse[(size_t)rc * D + d];
    sA[r * 257 + d] = 0.5f * x * (1.0f + erff(x * 0.70710678118f));
  }
  __syncthreads();

  { // logits = h @ gate_w2 + b2
    int pl = tid >> 2, e0 = (tid & 3) * 8;
    float a[8];
#pragma unroll
    for (int e = 0; e < 8; ++e) a[e] = gb2[e0 + e];
    for (int d = 0; d < D; ++d) {
      float hv = sA[pl * 257 + d];
#pragma unroll
      for (int e = 0; e < 8; ++e) a[e] += hv * gw2[(size_t)d * E + e0 + e];
    }
#pragma unroll
    for (int e = 0; e < 8; ++e) lg[pl * E + e0 + e] = a[e];
  }
  __syncthreads();

  { // Phase D: top-2 masked softmax + expert combine
    int pl = tid >> 2, d0 = (tid & 3) * 64;
    int pg = p0 + pl;
    if (pg < P) {
      float m1 = -INFINITY, m2 = -INFINITY;
      for (int e = 0; e < E; ++e) {
        float v = lg[pl * E + e];
        if (v > m1) { m2 = m1; m1 = v; } else if (v > m2) { m2 = v; }
      }
      float denom = 0.f;
      for (int e = 0; e < E; ++e) {
        float v = lg[pl * E + e];
        if (v >= m2) denom += __expf(v - m1);
      }
      float accv[64];
#pragma unroll
      for (int dd = 0; dd < 64; ++dd) accv[dd] = 0.f;
      for (int e = 0; e < E; ++e) {
        float v = lg[pl * E + e];
        if (v >= m2) {
          float wgt = __expf(v - m1) / denom;
          const float* qp = qexp + ((size_t)e * MAXP + pg) * D + d0;
#pragma unroll
          for (int dd = 0; dd < 64; dd += 4) {
            float4 q4 = *(const float4*)(qp + dd);
            accv[dd] += wgt * q4.x; accv[dd + 1] += wgt * q4.y;
            accv[dd + 2] += wgt * q4.z; accv[dd + 3] += wgt * q4.w;
          }
        }
      }
      float* op = out + ((size_t)b * P + pg) * D + d0;
#pragma unroll
      for (int dd = 0; dd < 64; dd += 4) {
        float4 q4 = {accv[dd], accv[dd + 1], accv[dd + 2], accv[dd + 3]};
        *(float4*)(op + dd) = q4;
      }
    }
  }
}

// ---------------------------------------------------------------------------
extern "C" void kernel_launch(void* const* d_in, const int* in_sizes, int n_in,
                              void* d_out, int out_size, void* d_ws, size_t ws_size,
                              hipStream_t stream) {
  const float* ctx  = (const float*)d_in[0];
  const float* qexp = (const float*)d_in[1];
  const float* qpos = (const float*)d_in[2];
  const float* lemb = (const float*)d_in[3];
  const float* lat  = (const float*)d_in[4];
  const float* wlq  = (const float*)d_in[5];
  const float* wck  = (const float*)d_in[6];
  const float* wcv  = (const float*)d_in[7];
  const float* wlo  = (const float*)d_in[8];
  const float* wsq  = (const float*)d_in[9];
  const float* wlk  = (const float*)d_in[10];
  const float* wlv  = (const float*)d_in[11];
  const float* wso  = (const float*)d_in[12];
  const float* gw1  = (const float*)d_in[13];
  const float* gb1  = (const float*)d_in[14];
  const float* gw2  = (const float*)d_in[15];
  const float* gb2  = (const float*)d_in[16];
  float* ws   = (float*)d_ws;
  float* outp = (float*)d_out;

  k0a<<<P + L + 2 * D, 256, 0, stream>>>(qpos, lemb, lat, wlq, wcv, wlo, wso, gw1, ws);
  k0b<<<L + 2 * P, 256, 0, stream>>>(wck, wsq, gw1, gb1, ws);
  k1<<<B * 2, 256, 0, stream>>>(ctx, ws);
  k2<<<B, 256, 0, stream>>>(wlk, wlv, ws);
  k3<<<dim3(3, B), 256, 0, stream>>>(ws, gw2, gb2, qexp, outp);
}
